// DynamicRoutingRNN_11622181503023
// MI455X (gfx1250) — compile-verified, matrix operand order checked by reading, not run on hardware
//
#include <hip/hip_runtime.h>
#include <hip/hip_bf16.h>

// ---------------- problem constants ----------------
#define RNN_N   200
#define RNN_NI  7
#define RNN_B   256
#define RNN_T   1000
#define BT      16        // batch tile per workgroup
#define NWAVE   7         // 224 threads; wave w owns cols [32w, 32w+32)
#define LDSTR   232       // bf16 elems per LDS row: 116 words -> conflict-free b128 loads

typedef __attribute__((ext_vector_type(16))) __bf16 v16bf;
typedef __attribute__((ext_vector_type(8)))  float  v8f;
typedef __attribute__((ext_vector_type(4)))  float  f32x4;

union AFrag {
    v16bf  v;
    f32x4  f4[2];
    __bf16 e[16];
};

struct Lane {
    int    tid, wv, half, l15, bstart;
    size_t nbase[2];   // noise element-index base (clamped col, per-lane invariant)
    size_t ybase[2];   // y_seq element-index base (true col; deref'd only when valid)
};

// LDS-only workgroup barrier: waits DScnt (LDS visibility) but lets global
// stores (y_seq) and prefetch loads (noise) stay in flight across the barrier.
__device__ __forceinline__ void lds_sync()
{
    asm volatile("s_wait_dscnt 0x0\n\t"
                 "s_barrier_signal -1\n\t"
                 "s_barrier_wait -1" ::: "memory");
}

__device__ __forceinline__ void load_afrag(AFrag& a, const __bf16* arow, int kc, int half)
{
    const int kb = 32 * kc + 8 * half;
    a.f4[0] = *reinterpret_cast<const f32x4*>(arow + kb);
    a.f4[1] = *reinterpret_cast<const f32x4*>(arow + kb + 16);
}

template<bool EDGE>
__device__ __forceinline__ void prefetch_noise(const float* __restrict__ noise,
                                               const Lane& L, int t, float (&nz)[2][8])
{
    const size_t TN = (size_t)RNN_T * RNN_N;
    #pragma unroll
    for (int tb = 0; tb < (EDGE ? 1 : 2); ++tb) {
        const float* np = noise + L.nbase[tb] + (size_t)t * RNN_N;  // fresh derivation
        #pragma unroll
        for (int i = 0; i < 8; ++i)
            nz[tb][i] = np[(size_t)i * TN];
    }
}

// One scan step: read LDS buffer P, write buffer P^1. Compile-time parity & wave class.
// nz is single-buffered: consumed for step t, then refilled for step t+1.
template<int P, bool EDGE>
__device__ __forceinline__ void rnn_step(
    int t,
    __bf16 (&yt)[2][BT][LDSTR],
    const AFrag (&Breg)[2][7],
    float (&yold)[2][8],
    float (&nz)[2][8],
    const float* __restrict__ noise,
    const float* __restrict__ u,
    float* __restrict__ y_seq,
    const Lane& L)
{
    constexpr int Q = P ^ 1;
    const size_t TN          = (size_t)RNN_T * RNN_N;
    const float  ALPHA       = 0.2f;
    const float  NOISE_SCALE = 0.15f * 0.632455532f;

    // ---- K loop: 4-deep pipelined A fetch, 14 (interior) / 7 (edge) WMMAs ----
    v8f acc0 = {}; v8f acc1 = {};
    const __bf16* arow = &yt[P][L.l15][0];
    AFrag a[4];
    #pragma unroll
    for (int kc = 0; kc < 4; ++kc) load_afrag(a[kc], arow, kc, L.half);
    #pragma unroll
    for (int kc = 0; kc < 7; ++kc) {
        acc0 = __builtin_amdgcn_wmma_f32_16x16x32_bf16(false, a[kc & 3].v, false,
                                                       Breg[0][kc].v, (short)0, acc0,
                                                       false, false);
        if (!EDGE)
            acc1 = __builtin_amdgcn_wmma_f32_16x16x32_bf16(false, a[kc & 3].v, false,
                                                           Breg[1][kc].v, (short)0, acc1,
                                                           false, false);
        if (kc + 4 < 7) load_afrag(a[kc & 3], arow, kc + 4, L.half);
    }

    // acc-independent decay terms: fills the WMMA->VALU hazard window
    float dec[2][8];
    #pragma unroll
    for (int tb = 0; tb < (EDGE ? 1 : 2); ++tb)
        #pragma unroll
        for (int i = 0; i < 8; ++i)
            dec[tb][i] = (1.0f - ALPHA) * yold[tb][i] + NOISE_SCALE * nz[tb][i];

    // ---- state update; write NEXT buffer (different buffer -> no barrier first) ----
    const size_t toff = (size_t)t * RNN_N;
    if (!EDGE) {
        // interior waves: both tiles fully valid, branch-free
        #pragma unroll
        for (int tb = 0; tb < 2; ++tb) {
            float* yp = y_seq + L.ybase[tb] + toff;        // one addr per tile per step
            const int n = 32 * L.wv + 16 * tb + L.l15;
            #pragma unroll
            for (int i = 0; i < 8; ++i) {
                const int m = i + 8 * L.half;
                const float pre = (tb == 0) ? acc0[i] : acc1[i];
                const float r   = pre > 0.0f ? pre : 0.0f;
                const float yn  = dec[tb][i] + ALPHA * r;
                yold[tb][i] = yn;
                yp[(size_t)i * TN] = yn;
                yt[Q][m][n] = (__bf16)yn;
            }
        }
    } else {
        // edge wave: tile 0 mixed (n = 192..207), tile 1 all-pad (skipped)
        const int n = 192 + L.l15;
        float ynv[8];
        #pragma unroll
        for (int i = 0; i < 8; ++i) {
            const float pre = acc0[i];
            const float r   = pre > 0.0f ? pre : 0.0f;
            ynv[i] = dec[0][i] + ALPHA * r;
            yold[0][i] = ynv[i];
        }
        if (n < RNN_N) {                                   // single guard, all stores inside
            float* yp = y_seq + L.ybase[0] + toff;
            #pragma unroll
            for (int i = 0; i < 8; ++i) {
                yp[(size_t)i * TN] = ynv[i];
                yt[Q][i + 8 * L.half][n] = (__bf16)ynv[i];
            }
        }
    }

    // stage u_{t+1} into write buffer (clamped index, no branch on t)
    if (L.tid < BT * RNN_NI) {
        const int tn = (t + 1 < RNN_T) ? (t + 1) : (RNN_T - 1);
        const int rr = L.tid / RNN_NI, ii = L.tid % RNN_NI;
        yt[Q][rr][RNN_N + ii] =
            (__bf16)u[(size_t)(L.bstart + rr) * RNN_T * RNN_NI + (size_t)tn * RNN_NI + ii];
    }

    // prefetch noise for the NEXT step; stays in flight across the LDS-only barrier
    prefetch_noise<EDGE>(noise, L, (t + 1 < RNN_T) ? (t + 1) : (RNN_T - 1), nz);

    lds_sync();   // LDS writes visible; global stores/loads continue in background
}

template<bool EDGE>
__device__ __forceinline__ void scan_loop(
    __bf16 (&yt)[2][BT][LDSTR],
    const AFrag (&Breg)[2][7],
    float (&yold)[2][8],
    const float* __restrict__ noise,
    const float* __restrict__ u,
    float* __restrict__ y_seq,
    const Lane& L)
{
    float nz[2][8];
    prefetch_noise<EDGE>(noise, L, 0, nz);
    lds_sync();

    #pragma unroll 1
    for (int t = 0; t < RNN_T; t += 2) {
        rnn_step<0, EDGE>(t,     yt, Breg, yold, nz, noise, u, y_seq, L);
        rnn_step<1, EDGE>(t + 1, yt, Breg, yold, nz, noise, u, y_seq, L);
    }
}

// ------------- persistent recurrent-scan kernel -------------
__global__ __launch_bounds__(NWAVE * 32, 1)
void rnn_scan_kernel(const float* __restrict__ y0,        // (B,N)
                     const float* __restrict__ u,         // (B,T,NI)
                     const float* __restrict__ noise,     // (B,T,N)
                     const float* __restrict__ W_in_raw,  // (N,NI)
                     const float* __restrict__ W_rec,     // (N,N)
                     const float* __restrict__ b_rec,     // (N)
                     float* __restrict__ y_seq,           // (B,T,N)
                     float* __restrict__ y_final)         // (B,N)
{
    __shared__ __align__(16) __bf16 yt[2][BT][LDSTR];   // ping-pong [y | u_t | 1 | 0] tile

    Lane L;
    L.tid    = threadIdx.x;
    const int lane = L.tid & 31;
    L.wv     = __builtin_amdgcn_readfirstlane(L.tid >> 5);  // scalar wave id
    L.half   = lane >> 4;
    L.l15    = lane & 15;
    L.bstart = blockIdx.x * BT;
    const size_t TN = (size_t)RNN_T * RNN_N;

    #pragma unroll
    for (int tb = 0; tb < 2; ++tb) {
        const int n  = 32 * L.wv + 16 * tb + L.l15;
        const int nc = (n < RNN_N) ? n : (RNN_N - 1);
        L.nbase[tb] = (size_t)(L.bstart + 8 * L.half) * TN + nc;
        L.ybase[tb] = (size_t)(L.bstart + 8 * L.half) * TN + n;
    }

    // ---- preload B fragments of Wtilde (time-invariant), vectorized ----
    // Wtilde[n][k]: k<200 -> W_rec[n][k]; 200..206 -> |W_in_raw[n][k-200]|;
    //               207 -> b_rec[n]; else 0.  Rows n>=200 zero.
    AFrag Breg[2][7];
    #pragma unroll
    for (int tb = 0; tb < 2; ++tb) {
        const int n = 32 * L.wv + 16 * tb + L.l15;
        if (n < RNN_N) {
            const float* wrow = W_rec + (size_t)n * RNN_N;
            #pragma unroll
            for (int kc = 0; kc < 6; ++kc) {          // K 0..191: pure W_rec, aligned b128
                const int kb = 32 * kc + 8 * L.half;
                const f32x4 r0 = *reinterpret_cast<const f32x4*>(wrow + kb);
                const f32x4 r1 = *reinterpret_cast<const f32x4*>(wrow + kb + 4);
                const f32x4 r2 = *reinterpret_cast<const f32x4*>(wrow + kb + 16);
                const f32x4 r3 = *reinterpret_cast<const f32x4*>(wrow + kb + 20);
                #pragma unroll
                for (int j = 0; j < 4; ++j) {
                    Breg[tb][kc].e[j]      = (__bf16)r0[j];
                    Breg[tb][kc].e[4 + j]  = (__bf16)r1[j];
                    Breg[tb][kc].e[8 + j]  = (__bf16)r2[j];
                    Breg[tb][kc].e[12 + j] = (__bf16)r3[j];
                }
            }
            #pragma unroll
            for (int j = 0; j < 16; ++j) {            // K 192..223: tail | u | bias | 0
                const int k = 192 + 8 * L.half + (j < 8 ? j : j + 8);
                float v = 0.0f;
                if (k < RNN_N)                      v = wrow[k];
                else if (k < RNN_N + RNN_NI)        v = fabsf(W_in_raw[n * RNN_NI + (k - RNN_N)]);
                else if (k == RNN_N + RNN_NI)       v = b_rec[n];
                Breg[tb][6].e[j] = (__bf16)v;
            }
        } else {
            #pragma unroll
            for (int kc = 0; kc < 7; ++kc)
                #pragma unroll
                for (int j = 0; j < 16; ++j) Breg[tb][kc].e[j] = (__bf16)0.0f;
        }
    }

    // ---- init f32 state registers + bf16 LDS tile (buffer 0) ----
    float yold[2][8];
    #pragma unroll
    for (int tb = 0; tb < 2; ++tb) {
        const int n = 32 * L.wv + 16 * tb + L.l15;
        #pragma unroll
        for (int i = 0; i < 8; ++i) {
            const int m = i + 8 * L.half;
            float v = 0.0f;
            if (n < RNN_N) {
                v = y0[(size_t)(L.bstart + m) * RNN_N + n];
                yt[0][m][n] = (__bf16)v;
            }
            yold[tb][i] = v;
        }
    }
    // bias (=1) + zero K-padding in BOTH buffers (k = 207..231)
    if (L.tid < BT) {
        #pragma unroll
        for (int buf = 0; buf < 2; ++buf)
            for (int k = RNN_N + RNN_NI; k < LDSTR; ++k)
                yt[buf][L.tid][k] = (k == RNN_N + RNN_NI) ? (__bf16)1.0f : (__bf16)0.0f;
    }
    // u columns for t = 0 into buffer 0
    if (L.tid < BT * RNN_NI) {
        const int r = L.tid / RNN_NI, i = L.tid % RNN_NI;
        yt[0][r][RNN_N + i] = (__bf16)u[(size_t)(L.bstart + r) * RNN_T * RNN_NI + i];
    }

    // ---- specialized scan: interior waves branch-free, edge wave handles the seam ----
    if (L.wv < 6) scan_loop<false>(yt, Breg, yold, noise, u, y_seq, L);
    else          scan_loop<true >(yt, Breg, yold, noise, u, y_seq, L);

    // ---- y_final from register state ----
    #pragma unroll
    for (int tb = 0; tb < 2; ++tb) {
        const int tbase = 32 * L.wv + 16 * tb;
        if (tbase < RNN_N) {
            const int n = tbase + L.l15;
            #pragma unroll
            for (int i = 0; i < 8; ++i) {
                const int m = i + 8 * L.half;
                if (n < RNN_N)
                    y_final[(size_t)(L.bstart + m) * RNN_N + n] = yold[tb][i];
            }
        }
    }
}

// ------------- streaming sigmoid readout: one wave per (b,t) -------------
__global__ __launch_bounds__(256)
void rnn_readout_kernel(const float* __restrict__ y_seq,   // (B*T, N)
                        const float* __restrict__ w_out,   // (N)
                        const float* __restrict__ b_out,   // (1)
                        float* __restrict__ z_seq)         // (B*T)
{
    const int lane = threadIdx.x & 31;
    const long long row = (long long)blockIdx.x * (blockDim.x >> 5) + (threadIdx.x >> 5);
    if (row >= (long long)RNN_B * RNN_T) return;

    float s = 0.0f;
    for (int n = lane; n < RNN_N; n += 32)
        s += y_seq[row * RNN_N + n] * w_out[n];

    #pragma unroll
    for (int off = 16; off > 0; off >>= 1)
        s += __shfl_xor(s, off, 32);

    if (lane == 0)
        z_seq[row] = 1.0f / (1.0f + __expf(-(s + b_out[0])));
}

extern "C" void kernel_launch(void* const* d_in, const int* in_sizes, int n_in,
                              void* d_out, int out_size, void* d_ws, size_t ws_size,
                              hipStream_t stream) {
    const float* y0       = (const float*)d_in[0];
    const float* u        = (const float*)d_in[1];
    const float* noise    = (const float*)d_in[2];
    const float* W_in_raw = (const float*)d_in[3];
    const float* W_rec    = (const float*)d_in[4];
    const float* b_rec    = (const float*)d_in[5];
    const float* w_out    = (const float*)d_in[6];
    const float* b_out    = (const float*)d_in[7];

    float* out     = (float*)d_out;
    float* y_seq   = out;                                  // B*T*N
    float* z_seq   = out + (size_t)RNN_B * RNN_T * RNN_N;  // B*T
    float* y_final = z_seq + (size_t)RNN_B * RNN_T;        // B*N

    rnn_scan_kernel<<<RNN_B / BT, NWAVE * 32, 0, stream>>>(
        y0, u, noise, W_in_raw, W_rec, b_rec, y_seq, y_final);

    const int rows = RNN_B * RNN_T;
    const int wavesPerBlock = 8;   // 256 threads
    const int blocks = (rows + wavesPerBlock - 1) / wavesPerBlock;
    rnn_readout_kernel<<<blocks, wavesPerBlock * 32, 0, stream>>>(
        y_seq, w_out, b_out, z_seq);
}
